// MoELayer_31009663877642
// MI455X (gfx1250) — matrix operand order repaired and checked by reading, not compiled, not run on hardware
//
#include <hip/hip_runtime.h>
#include <math.h>

typedef __attribute__((ext_vector_type(16))) __bf16 v16bf;
typedef __attribute__((ext_vector_type(8)))  float  v8f;

#define B_TOK 8192
#define D_IN  768
#define H_DIM 1024
#define E_EXP 8

__device__ __forceinline__ unsigned short f2bf(float f) {
  unsigned u = __float_as_uint(f);
  unsigned r = u + 0x7FFFu + ((u >> 16) & 1u);
  return (unsigned short)(r >> 16);
}

// ---------------------------------------------------------------------------
// Kernel 0: zero the MoE output accumulator and the routing counters
// ---------------------------------------------------------------------------
__global__ __launch_bounds__(256)
void moe_zero_kernel(float* out, int* counts, int n) {
  int i = blockIdx.x * blockDim.x + threadIdx.x;
  for (; i < n; i += gridDim.x * blockDim.x) out[i] = 0.0f;
  if (blockIdx.x == 0 && threadIdx.x < E_EXP) counts[threadIdx.x] = 0;
}

// ---------------------------------------------------------------------------
// Kernel 1: pack W1 [E,768,1024] and W2 [E,1024,768] (K-major) into bf16 in
// the WMMA 16x16x32 B-operand lane layout:
//   tile (nT,kT): lane<16 -> n = nT*16+lane,    k = kT*32 + (j<8 ? j : j+8)     (j = 0..15)
//                 lane>=16 -> n = nT*16+lane-16, k = kT*32 + 8 + (j<8 ? j : j+8)
// 32 contiguous bytes per lane per tile -> 128-bit vector loads in the GEMM.
// ---------------------------------------------------------------------------
__global__ __launch_bounds__(256)
void moe_pack_kernel(const float* __restrict__ W1, const float* __restrict__ W2,
                     unsigned short* __restrict__ p1, unsigned short* __restrict__ p2) {
  const int HALF1 = E_EXP * 64 * 24 * 512;   // 8*768*1024
  const int HALF2 = E_EXP * 48 * 32 * 512;   // 8*1024*768
  for (int i = blockIdx.x * blockDim.x + threadIdx.x; i < HALF1 + HALF2;
       i += gridDim.x * blockDim.x) {
    const float* W; unsigned short* dst; int f, kTiles, nTiles, K, N;
    if (i < HALF1) { W = W1; dst = p1; f = i; kTiles = 24; nTiles = 64; K = D_IN;  N = H_DIM; }
    else           { W = W2; dst = p2; f = i - HALF1; kTiles = 32; nTiles = 48; K = H_DIM; N = D_IN; }
    int j    = f & 15;
    int lane = (f >> 4) & 31;
    int r    = f >> 9;
    int kT   = r % kTiles;  r /= kTiles;
    int nT   = r % nTiles;
    int e    = r / nTiles;
    int n = nT * 16 + (lane & 15);
    int k = kT * 32 + ((lane < 16) ? 0 : 8) + ((j < 8) ? j : j + 8);
    dst[f] = f2bf(W[(size_t)e * K * N + (size_t)k * N + n]);
  }
}

// ---------------------------------------------------------------------------
// Kernel 2: noisy top-2 gating. One wave32 per token.
// ---------------------------------------------------------------------------
__global__ __launch_bounds__(256)
void moe_gate_kernel(const float* __restrict__ x, const float* __restrict__ noise,
                     const float* __restrict__ Wg, const float* __restrict__ bg,
                     const float* __restrict__ Wn, const float* __restrict__ bn,
                     float* __restrict__ clean_out, int* __restrict__ topidx_out,
                     int* __restrict__ counts, int* __restrict__ tokenList,
                     float* __restrict__ wList) {
  const int t    = blockIdx.x * 8 + (threadIdx.x >> 5);
  const int lane = threadIdx.x & 31;
  if (t >= B_TOK) return;

  float ag[E_EXP], an[E_EXP];
#pragma unroll
  for (int e = 0; e < E_EXP; e++) { ag[e] = 0.f; an[e] = 0.f; }

  for (int d = lane; d < D_IN; d += 32) {
    float xv = x[(size_t)t * D_IN + d];
    const float4* g4 = (const float4*)(Wg + d * E_EXP);
    const float4* n4 = (const float4*)(Wn + d * E_EXP);
    float4 g0 = g4[0], g1 = g4[1], m0 = n4[0], m1 = n4[1];
    ag[0] += xv * g0.x; ag[1] += xv * g0.y; ag[2] += xv * g0.z; ag[3] += xv * g0.w;
    ag[4] += xv * g1.x; ag[5] += xv * g1.y; ag[6] += xv * g1.z; ag[7] += xv * g1.w;
    an[0] += xv * m0.x; an[1] += xv * m0.y; an[2] += xv * m0.z; an[3] += xv * m0.w;
    an[4] += xv * m1.x; an[5] += xv * m1.y; an[6] += xv * m1.z; an[7] += xv * m1.w;
  }
#pragma unroll
  for (int off = 16; off > 0; off >>= 1) {
#pragma unroll
    for (int e = 0; e < E_EXP; e++) {
      ag[e] += __shfl_xor(ag[e], off, 32);
      an[e] += __shfl_xor(an[e], off, 32);
    }
  }

  if (lane == 0) {
    float noisy[E_EXP];
#pragma unroll
    for (int e = 0; e < E_EXP; e++) {
      float clean = ag[e] + bg[e];
      clean_out[(size_t)t * E_EXP + e] = clean;
      float z  = an[e] + bn[e];
      float sp = (z > 20.f) ? z : log1pf(expf(z));          // softplus
      noisy[e] = clean + noise[(size_t)t * E_EXP + e] * sp;
    }
    int i1 = 0; float v1 = noisy[0];
#pragma unroll
    for (int e = 1; e < E_EXP; e++) if (noisy[e] > v1) { v1 = noisy[e]; i1 = e; }
    int i2 = (i1 == 0) ? 1 : 0; float v2 = noisy[i2];
#pragma unroll
    for (int e = 0; e < E_EXP; e++)
      if (e != i1 && e != i2 && noisy[e] > v2) { v2 = noisy[e]; i2 = e; }
    float e2 = expf(v2 - v1);
    float inv = 1.f / (1.f + e2);
    float w1 = inv, w2 = e2 * inv;
    topidx_out[t * 2]     = i1;
    topidx_out[t * 2 + 1] = i2;
    int p1 = atomicAdd(&counts[i1], 1);
    tokenList[i1 * B_TOK + p1] = t; wList[i1 * B_TOK + p1] = w1;
    int p2 = atomicAdd(&counts[i2], 1);
    tokenList[i2 * B_TOK + p2] = t; wList[i2 * B_TOK + p2] = w2;
  }
}

// ---------------------------------------------------------------------------
// Kernel 3: fused expert FFN. Block = (32 routed tokens, one expert),
// 512 threads = 16 wave32 (4 waves/SIMD for latency hiding).
// Per wave: 4 N-tiles x 2 M-halves of accumulators (64 VGPRs), with an
// explicit ping-pong double buffer of B tiles so the next K-step's global
// loads are in flight while the current K-step's WMMAs execute.
// x tile gathered with GLOBAL_LOAD_ASYNC_TO_LDS_B128 (ASYNCcnt path).
// ---------------------------------------------------------------------------
__global__ __launch_bounds__(512)
void moe_expert_kernel(const float* __restrict__ x, const float* __restrict__ b1,
                       const float* __restrict__ lng, const float* __restrict__ lnb,
                       const float* __restrict__ b2, const int* __restrict__ counts,
                       const int* __restrict__ tokenList, const float* __restrict__ wList,
                       const unsigned short* __restrict__ pW1,
                       const unsigned short* __restrict__ pW2,
                       float* __restrict__ out) {
  const int e    = blockIdx.y;
  const int tile = blockIdx.x;
  const int cnt  = counts[e];
  if (tile * 32 >= cnt) return;   // uniform across block

  extern __shared__ __align__(32) char smem[];
  // [0, 64K):     Apack — bf16 A-operand, [kStep][mHalf][lane][16]
  // [64K, 192K):  overlay: rawX (32x768 f32, 96K) then Hbuf (32x1024 f32, 128K)
  // [192K, ...):  LN partials + row params
  unsigned short* Apack = (unsigned short*)smem;
  float* rawX = (float*)(smem + 65536);
  float* Hbuf = (float*)(smem + 65536);
  float* P    = (float*)(smem + 65536 + 131072);
  float* Q    = P + 512;
  float* Mu   = Q + 512;
  float* Rs   = Mu + 32;
  float* wgtS = Rs + 32;
  int*   tokS = (int*)(wgtS + 32);

  const int tid  = threadIdx.x;
  const int lane = tid & 31;
  const int w    = tid >> 5;     // 0..15

  if (tid < 32) {
    int idx = tile * 32 + tid;
    if (idx < cnt) {
      tokS[tid] = tokenList[e * B_TOK + idx];
      wgtS[tid] = wList[e * B_TOK + idx];
    } else {
      tokS[tid] = 0; wgtS[tid] = 0.f;   // padded row contributes 0
    }
  }
  __syncthreads();

  // --- async gather of the 32 routed x rows into LDS (raw fp32) -------------
  for (int f = tid; f < 32 * 192; f += 512) {
    int m  = f / 192;
    int c4 = f % 192;                       // 4-float chunk within the row
    unsigned long long gaddr =
        (unsigned long long)(const void*)(x + (size_t)tokS[m] * D_IN + c4 * 4);
    unsigned ldsoff = (unsigned)(size_t)(const void*)(rawX + m * D_IN + c4 * 4);
    asm volatile("global_load_async_to_lds_b128 %0, %1, off"
                 :: "v"(ldsoff), "v"(gaddr) : "memory");
  }
  asm volatile("s_wait_asynccnt 0x0" ::: "memory");
  __syncthreads();

  // --- pack rawX -> bf16 WMMA A-operand layout (K = D, 24 K-steps, 2 halves)
  for (int f = tid; f < 24 * 1024; f += 512) {
    int j  = f & 15;
    int ln = (f >> 4) & 31;
    int mh = (f >> 9) & 1;
    int ks = f >> 10;
    int m  = mh * 16 + (ln & 15);
    int kk = (ln < 16) ? ((j < 8) ? j : j + 8) : ((j < 8) ? j + 8 : j + 16);
    Apack[f] = f2bf(rawX[m * D_IN + ks * 32 + kk]);
  }
  __syncthreads();

  // --- GEMM1: h[32,1024] = xA @ W1[e]; wave w owns N-tiles w*4 .. w*4+3,
  //     both M halves. Explicit double-buffered B tiles.
  v8f zf = {0.f, 0.f, 0.f, 0.f, 0.f, 0.f, 0.f, 0.f};
  v8f acc[8];
#pragma unroll
  for (int i = 0; i < 8; i++) acc[i] = zf;
  const v16bf* Bp1 = (const v16bf*)pW1;

  auto loadB1 = [&](v16bf (&bb)[4], int ks) {
#pragma unroll
    for (int nt = 0; nt < 4; nt++)
      bb[nt] = Bp1[((e * 64 + (w * 4 + nt)) * 24 + ks) * 32 + lane];
  };
  auto mmA = [&](v16bf (&bb)[4], int ks) {
    v16bf a0 = *(const v16bf*)(Apack + (ks * 2 + 0) * 512 + lane * 16);
    v16bf a1 = *(const v16bf*)(Apack + (ks * 2 + 1) * 512 + lane * 16);
#pragma unroll
    for (int nt = 0; nt < 4; nt++) {
      acc[nt * 2 + 0] = __builtin_amdgcn_wmma_f32_16x16x32_bf16(
          false, a0, false, bb[nt], (short)0, acc[nt * 2 + 0], false, false);
      acc[nt * 2 + 1] = __builtin_amdgcn_wmma_f32_16x16x32_bf16(
          false, a1, false, bb[nt], (short)0, acc[nt * 2 + 1], false, false);
    }
  };

  {
    v16bf bA[4], bB[4];
    loadB1(bA, 0);
#pragma unroll
    for (int ks2 = 0; ks2 < 12; ks2++) {
      const int ks0 = 2 * ks2;
      loadB1(bB, ks0 + 1);            // in flight while bA is consumed
      mmA(bA, ks0);
      if (ks2 < 11) loadB1(bA, ks0 + 2);
      mmA(bB, ks0 + 1);
    }
  }

  // rawX is dead; Hbuf overlays it. Write h + b1 to Hbuf.
  {
    int m0 = (lane < 16) ? 0 : 8;
    int nn = lane & 15;
#pragma unroll
    for (int nt = 0; nt < 4; nt++) {
      int n = (w * 4 + nt) * 16 + nn;
      float bias = b1[e * H_DIM + n];
#pragma unroll
      for (int mh = 0; mh < 2; mh++)
#pragma unroll
        for (int r = 0; r < 8; r++)
          Hbuf[(mh * 16 + m0 + r) * H_DIM + n] = acc[nt * 2 + mh][r] + bias;
    }
  }
  __syncthreads();

  // --- LayerNorm stats over H=1024 per row (16 threads per row, 32 rows)
  {
    int m = tid >> 4, part = tid & 15;
    float s = 0.f, sq = 0.f;
    for (int i = 0; i < 64; i++) {
      float v = Hbuf[m * H_DIM + part * 64 + i];
      s += v; sq += v * v;
    }
    P[m * 16 + part] = s; Q[m * 16 + part] = sq;
  }
  __syncthreads();
  if (tid < 32) {
    float s = 0.f, sq = 0.f;
    for (int i = 0; i < 16; i++) { s += P[tid * 16 + i]; sq += Q[tid * 16 + i]; }
    float mu  = s * (1.f / 1024.f);
    float var = sq * (1.f / 1024.f) - mu * mu;
    Mu[tid] = mu; Rs[tid] = rsqrtf(var + 1e-5f);
  }
  __syncthreads();

  // --- LN + exact GELU, repack to bf16 A-operand layout (K = H, 32 K-steps)
  for (int f = tid; f < 32 * 1024; f += 512) {
    int j  = f & 15;
    int ln = (f >> 4) & 31;
    int mh = (f >> 9) & 1;
    int ks = f >> 10;
    int m  = mh * 16 + (ln & 15);
    int kk = (ln < 16) ? ((j < 8) ? j : j + 8) : ((j < 8) ? j + 8 : j + 16);
    int k  = ks * 32 + kk;
    float v = Hbuf[m * H_DIM + k];
    v = (v - Mu[m]) * Rs[m] * lng[e * H_DIM + k] + lnb[e * H_DIM + k];
    v = 0.5f * v * (1.f + erff(v * 0.70710678118654752f));   // exact GELU
    Apack[f] = f2bf(v);
  }
  __syncthreads();

  // --- GEMM2: out[32,768] = gelu(h) @ W2[e]; wave w owns N-tiles w*3..w*3+2
  v8f acc2[6];
#pragma unroll
  for (int i = 0; i < 6; i++) acc2[i] = zf;
  const v16bf* Bp2 = (const v16bf*)pW2;

  auto loadB2 = [&](v16bf (&bb)[3], int ks) {
#pragma unroll
    for (int nt = 0; nt < 3; nt++)
      bb[nt] = Bp2[((e * 48 + (w * 3 + nt)) * 32 + ks) * 32 + lane];
  };
  auto mmB = [&](v16bf (&bb)[3], int ks) {
    v16bf a0 = *(const v16bf*)(Apack + (ks * 2 + 0) * 512 + lane * 16);
    v16bf a1 = *(const v16bf*)(Apack + (ks * 2 + 1) * 512 + lane * 16);
#pragma unroll
    for (int nt = 0; nt < 3; nt++) {
      acc2[nt * 2 + 0] = __builtin_amdgcn_wmma_f32_16x16x32_bf16(
          false, a0, false, bb[nt], (short)0, acc2[nt * 2 + 0], false, false);
      acc2[nt * 2 + 1] = __builtin_amdgcn_wmma_f32_16x16x32_bf16(
          false, a1, false, bb[nt], (short)0, acc2[nt * 2 + 1], false, false);
    }
  };

  {
    v16bf cA[3], cB[3];
    loadB2(cA, 0);
#pragma unroll
    for (int ks2 = 0; ks2 < 16; ks2++) {
      const int ks0 = 2 * ks2;
      loadB2(cB, ks0 + 1);
      mmB(cA, ks0);
      if (ks2 < 15) loadB2(cA, ks0 + 2);
      mmB(cB, ks0 + 1);
    }
  }

  // --- gate-weighted scatter-accumulate into d_out
  {
    int m0 = (lane < 16) ? 0 : 8;
    int nn = lane & 15;
#pragma unroll
    for (int nt = 0; nt < 3; nt++) {
      int n = (w * 3 + nt) * 16 + nn;
      float bias = b2[e * D_IN + n];
#pragma unroll
      for (int mh = 0; mh < 2; mh++)
#pragma unroll
        for (int r = 0; r < 8; r++) {
          int m = mh * 16 + m0 + r;
          float v = (acc2[nt * 2 + mh][r] + bias) * wgtS[m];
          atomicAdd(&out[(size_t)tokS[m] * D_IN + n], v);
        }
    }
  }
}

// ---------------------------------------------------------------------------
extern "C" void kernel_launch(void* const* d_in, const int* in_sizes, int n_in,
                              void* d_out, int out_size, void* d_ws, size_t ws_size,
                              hipStream_t stream) {
  const float* x     = (const float*)d_in[0];
  const float* noise = (const float*)d_in[1];
  const float* Wg    = (const float*)d_in[2];
  const float* bg    = (const float*)d_in[3];
  const float* Wn    = (const float*)d_in[4];
  const float* bn    = (const float*)d_in[5];
  const float* W1    = (const float*)d_in[6];
  const float* b1    = (const float*)d_in[7];
  const float* lng   = (const float*)d_in[8];
  const float* lnb   = (const float*)d_in[9];
  const float* W2    = (const float*)d_in[10];
  const float* b2    = (const float*)d_in[11];

  float* out       = (float*)d_out;                             // [B, D]
  float* clean_out = out + (size_t)B_TOK * D_IN;                // [B, E]
  int*   topidx    = (int*)(clean_out + (size_t)B_TOK * E_EXP); // [B, 2]

  char* ws = (char*)d_ws;
  int*   counts    = (int*)ws;                                  // 32 B
  int*   tokenList = (int*)(ws + 32);                           // 256 KB
  float* wList     = (float*)(ws + 32 + 4 * E_EXP * B_TOK);     // 256 KB
  unsigned short* pW1 = (unsigned short*)(ws + 32 + 8 * E_EXP * B_TOK);
  unsigned short* pW2 = pW1 + (size_t)E_EXP * D_IN * H_DIM;

  moe_zero_kernel<<<4096, 256, 0, stream>>>(out, counts, B_TOK * D_IN);
  moe_pack_kernel<<<8192, 256, 0, stream>>>(W1, W2, pW1, pW2);
  moe_gate_kernel<<<B_TOK / 8, 256, 0, stream>>>(x, noise, Wg, bg, Wn, bn,
                                                 clean_out, topidx,
                                                 counts, tokenList, wList);
  size_t smem = 65536 + 131072 + (512 + 512 + 32 + 32 + 32 + 32) * sizeof(float);
  moe_expert_kernel<<<dim3(B_TOK / 32, E_EXP), 512, smem, stream>>>(
      x, b1, lng, lnb, b2, counts, tokenList, wList, pW1, pW2, out);
}
